// FISA_24824910971306
// MI455X (gfx1250) — matrix operand order repaired and checked by reading, not compiled, not run on hardware
//
#include <hip/hip_runtime.h>
#include <math.h>

// ---------------------------------------------------------------------------
// Dims
#define Tn   8192
#define Dn   5
#define Fn   768
#define Hn   64
#define G4n  256          // 4*H
#define DTn  40960        // Dn*Tn

typedef __attribute__((ext_vector_type(16))) __bf16        v16bf;
typedef __attribute__((ext_vector_type(8)))  float         v8f;
typedef __attribute__((ext_vector_type(4)))  float         fvec4;
typedef __attribute__((ext_vector_type(8)))  unsigned int  uivec8;

union BFrag { v16bf bf; uivec8 ui; unsigned int u[8]; };

static __device__ __forceinline__ v8f zero8() {
  v8f z; for (int i = 0; i < 8; ++i) z[i] = 0.f; return z;
}

static __device__ __forceinline__ unsigned short f2bf16(float x) {
  unsigned int u = __float_as_uint(x);
  u = (u + 0x7FFFu + ((u >> 16) & 1u)) >> 16;     // round-to-nearest-even
  return (unsigned short)u;
}

static __device__ __forceinline__ unsigned int pack_bf(float a, float b) {
  return (unsigned int)f2bf16(a) | ((unsigned int)f2bf16(b) << 16);
}

static __device__ __forceinline__ float fast_sig(float x)  { return 1.f / (1.f + __expf(-x)); }
static __device__ __forceinline__ float fast_tanh(float x) { return 1.f - 2.f / (__expf(2.f * x) + 1.f); }

static __device__ __forceinline__ v8f wmma_bf16(const BFrag& a, const BFrag& b, v8f c) {
  return __builtin_amdgcn_wmma_f32_16x16x32_bf16(false, a.bf, false, b.bf, (short)0, c, false, false);
}

// Pack fp32 run [p0 p1 | p2 p3] (two 8-float runs) into an A/B fragment half.
static __device__ __forceinline__ void pack8(BFrag& f, fvec4 p0, fvec4 p1, fvec4 p2, fvec4 p3) {
  f.u[0] = pack_bf(p0[0], p0[1]); f.u[1] = pack_bf(p0[2], p0[3]);
  f.u[2] = pack_bf(p1[0], p1[1]); f.u[3] = pack_bf(p1[2], p1[3]);
  f.u[4] = pack_bf(p2[0], p2[1]); f.u[5] = pack_bf(p2[2], p2[3]);
  f.u[6] = pack_bf(p3[0], p3[1]); f.u[7] = pack_bf(p3[2], p3[3]);
}

// ---------------------------------------------------------------------------
// 1) Pack W matrices (row-major [N,K], y = x@W^T) into bf16 B-fragments.
//    B 32x16 bf16 layout: lanes0-15 K=0..7 & 16..23 (2/VGPR), lanes16-31 K=8..15 & 24..31, col=lane%16.
//    Buffer layout: frag(kc,nt): dword index = ((kc*NT + nt)*32 + lane)*8 + v.
__global__ __launch_bounds__(256)
void pack_weights(const float* __restrict__ Uw, const float* __restrict__ Ww,
                  const float* __restrict__ Wdw,
                  unsigned int* __restrict__ pbU, unsigned int* __restrict__ pbW,
                  unsigned int* __restrict__ pbWd)
{
  int gid = blockIdx.x * 256 + threadIdx.x;
  if (gid < 98304) {                       // U: 24 kchunks x 16 ntiles
    int v = gid & 7, lane = (gid >> 3) & 31, nt = (gid >> 8) & 15, kc = gid >> 12;
    int n = nt * 16 + (lane & 15);
    int k = kc * 32 + (lane >> 4) * 8 + (v >> 2) * 16 + 2 * (v & 3);
    pbU[gid] = pack_bf(Uw[n * Fn + k], Uw[n * Fn + k + 1]);
  } else if (gid < 98304 + 8192) {         // W_all: 2 kchunks x 16 ntiles
    int id = gid - 98304;
    int v = id & 7, lane = (id >> 3) & 31, nt = (id >> 8) & 15, kc = id >> 12;
    int n = nt * 16 + (lane & 15);
    int k = kc * 32 + (lane >> 4) * 8 + (v >> 2) * 16 + 2 * (v & 3);
    pbW[id] = pack_bf(Ww[n * Hn + k], Ww[n * Hn + k + 1]);
  } else if (gid < 98304 + 8192 + 2048) {  // W_d: 2 kchunks x 4 ntiles
    int id = gid - 98304 - 8192;
    int v = id & 7, lane = (id >> 3) & 31, nt = (id >> 8) & 3, kc = id >> 10;
    int n = nt * 16 + (lane & 15);
    int k = kc * 32 + (lane >> 4) * 8 + (v >> 2) * 16 + 2 * (v & 3);
    pbWd[id] = pack_bf(Wdw[n * Hn + k], Wdw[n * Hn + k + 1]);
  }
}

// ---------------------------------------------------------------------------
// 2) ux[t][d][g] = x[d,t,:] @ U^T + (U_b + W_all_b), via bf16 WMMA.
//    Grid (4, 320): blockIdx.x = 64-col group, blockIdx.y = 128-row group.
//    Wave w owns one 16-row M-tile and 4 N-tiles.
__global__ __launch_bounds__(256)
void ux_gemm(const float* __restrict__ x, const unsigned int* __restrict__ pbU,
             const float* __restrict__ Ub, const float* __restrict__ Wb,
             float* __restrict__ ux)
{
  const int w = threadIdx.x >> 5, lane = threadIdx.x & 31;
  const int m0 = blockIdx.y * 128 + w * 16;
  const int n0 = blockIdx.x * 64;

  v8f acc[4];
  for (int j = 0; j < 4; ++j) acc[j] = zero8();

  const int r    = lane & 15;
  const int koff = (lane >> 4) * 8;   // A layout: lanes0-15 K 0-7 & 16-23; lanes16-31 K 8-15 & 24-31
  const float* arow = x + (size_t)(m0 + r) * Fn + koff;

  for (int kc = 0; kc < 24; ++kc) {
    const float* ap = arow + kc * 32;
    fvec4 p0 = *(const fvec4*)(ap);
    fvec4 p1 = *(const fvec4*)(ap + 4);
    fvec4 p2 = *(const fvec4*)(ap + 16);
    fvec4 p3 = *(const fvec4*)(ap + 20);
    BFrag A; pack8(A, p0, p1, p2, p3);
    for (int j = 0; j < 4; ++j) {
      int ntile = (n0 >> 4) + j;
      BFrag B;
      B.ui = *(const uivec8*)(pbU + ((size_t)(kc * 16 + ntile) * 32 + lane) * 8);
      acc[j] = wmma_bf16(A, B, acc[j]);
    }
  }
  const int rbase = (lane >> 4) * 8;
  for (int j = 0; j < 4; ++j) {
    int col = n0 + j * 16 + (lane & 15);
    float bias = Ub[col] + Wb[col];
    for (int v = 0; v < 8; ++v) {
      int m = m0 + rbase + v;
      int d = m >> 13, t = m & 8191;
      ux[(size_t)t * (Dn * G4n) + d * G4n + col] = acc[j][v] + bias;
    }
  }
}

// ---------------------------------------------------------------------------
// 3) Sequential TimeLSTM scan: single workgroup (8 waves).
//    Key trick: the state-update phase writes h and c into LDS already in
//    bf16 A-fragment layout (16-bit LDS stores), so the per-step pack is just
//    two b256 LDS loads per wave — no conversion VALU on the serial path.
__global__ __launch_bounds__(256)
void time_lstm_scan(const float* __restrict__ ux, const float* __restrict__ tim,
                    const unsigned int* __restrict__ pbW, const unsigned int* __restrict__ pbWd,
                    const float* __restrict__ Wdb,
                    float* __restrict__ y, float* __restrict__ hfin)
{
  __shared__ uivec8 hpack[2 * 32];     // bf16 A-frag staged h: [kchunk][lane] x 8 dw
  __shared__ uivec8 cpack[2 * 32];     // bf16 A-frag staged c
  __shared__ float  cbuf[Dn * Hn];     // fp32 c (for c_adj)
  __shared__ float  hbuf[Dn * Hn];     // fp32 h (for h_fin)
  __shared__ float  G[4 * Dn * Hn];    // f,i,o,tanh(g) after activation
  __shared__ float  CADJ[Dn * Hn];
  __shared__ float  wdb[Hn];

  const int tid = threadIdx.x, w = tid >> 5, lane = tid & 31;
  { // zero state
    unsigned int* hz = (unsigned int*)hpack;
    unsigned int* cz = (unsigned int*)cpack;
    for (int i = tid; i < 512; i += 256) { hz[i] = 0u; cz[i] = 0u; }
    for (int i = tid; i < Dn * Hn; i += 256) { cbuf[i] = 0.f; hbuf[i] = 0.f; }
    if (tid < Hn) wdb[tid] = Wdb[tid];
  }

  BFrag Bg[2][2];                      // [kchunk][ntile-within-wave]
  for (int kc = 0; kc < 2; ++kc)
    for (int j = 0; j < 2; ++j)
      Bg[kc][j].ui = *(const uivec8*)(pbW + ((size_t)(kc * 16 + 2 * w + j) * 32 + lane) * 8);
  BFrag Bd[2];
  if (w < 4)
    for (int kc = 0; kc < 2; ++kc)
      Bd[kc].ui = *(const uivec8*)(pbWd + ((size_t)(kc * 4 + w) * 32 + lane) * 8);
  __syncthreads();

  const int gt = w >> 1;               // 0=f 1=i 2=o 3=g
  unsigned short* hs = (unsigned short*)hpack;
  unsigned short* cs = (unsigned short*)cpack;

  for (int t = 0; t < Tn; ++t) {
    if (t + 1 < Tn) __builtin_prefetch(ux + (size_t)(t + 1) * (Dn * G4n) + tid * 5, 0, 1);

    // loads that do NOT depend on the recurrence -> issued first
    float uxv[2][Dn];
    if (lane < 16) {
      for (int j = 0; j < 2; ++j) {
        int col = w * 32 + j * 16 + lane;
        const float* up = ux + (size_t)t * (Dn * G4n) + col;
        for (int v = 0; v < Dn; ++v) uxv[j][v] = up[v * G4n];
      }
    }
    float tdm1[Dn];
    for (int v = 0; v < Dn; ++v) tdm1[v] = tim[v * Tn + t] - 1.f;

    // A fragments: already bf16-packed in LDS
    BFrag Ah[2];
    Ah[0].ui = hpack[lane];
    Ah[1].ui = hpack[32 + lane];

    v8f accg[2]; accg[0] = zero8(); accg[1] = zero8();
    for (int j = 0; j < 2; ++j) {
      accg[j] = wmma_bf16(Ah[0], Bg[0][j], accg[j]);
      accg[j] = wmma_bf16(Ah[1], Bg[1][j], accg[j]);
    }

    // waves 0-3: c_s1 = tanh(c @ W_d^T + b)  ->  c_adj
    if (w < 4) {
      BFrag Ac[2];
      Ac[0].ui = cpack[lane];
      Ac[1].ui = cpack[32 + lane];
      v8f accc = zero8();
      accc = wmma_bf16(Ac[0], Bd[0], accc);
      accc = wmma_bf16(Ac[1], Bd[1], accc);
      if (lane < 16) {
        int col = w * 16 + lane;
        float wb = wdb[col];
        for (int v = 0; v < Dn; ++v) {
          float cv  = cbuf[v * Hn + col];
          float cs1 = fast_tanh(accc[v] + wb);
          CADJ[v * Hn + col] = cv + cs1 * tdm1[v];
        }
      }
    }

    // activations on the 5 valid rows, scatter to gate buffers
    if (lane < 16) {
      for (int j = 0; j < 2; ++j) {
        int cidx = (w & 1) * 32 + j * 16 + lane;
        for (int v = 0; v < Dn; ++v) {
          float val = accg[j][v] + uxv[j][v];
          float a   = (gt == 3) ? fast_tanh(val) : fast_sig(val);
          G[(gt * Dn + v) * Hn + cidx] = a;
        }
      }
    }
    __syncthreads();

    // state update; write next h/c directly in bf16 A-fragment layout
    if (tid < Dn * Hn) {
      int d = tid >> 6, k = tid & 63;
      float f  = G[(0 * Dn + d) * Hn + k];
      float i  = G[(1 * Dn + d) * Hn + k];
      float o  = G[(2 * Dn + d) * Hn + k];
      float tg = G[(3 * Dn + d) * Hn + k];
      float cn = f * CADJ[d * Hn + k] + i * tg;
      float hn = o * fast_tanh(cn);
      cbuf[d * Hn + k] = cn;
      hbuf[d * Hn + k] = hn;
      // A-frag address for element (row=d, K=k):
      int kc   = k >> 5, kk = k & 31;
      int vv   = ((kk >> 4) & 1) * 4 + ((kk & 7) >> 1);
      int ln   = d + 16 * ((kk >> 3) & 1);
      int si   = ((kc * 32 + ln) * 8 + vv) * 2 + (kk & 1);
      hs[si] = f2bf16(hn);
      cs[si] = f2bf16(cn);
      y[((size_t)d * Tn + t) * Hn + k] = fast_tanh(o);
    }
    __syncthreads();
  }
  if (tid < Dn * Hn) hfin[tid] = hbuf[tid];
}

// ---------------------------------------------------------------------------
// 4) q[d] = h_fin[d] @ W1^T + b1
__global__ __launch_bounds__(64)
void q_kernel(const float* __restrict__ hfin, const float* __restrict__ W1,
              const float* __restrict__ b1, float* __restrict__ q)
{
  int k = threadIdx.x;
  for (int d = 0; d < Dn; ++d) {
    float s = b1[k];
    for (int j = 0; j < Hn; ++j) s += hfin[d * Hn + j] * W1[k * Hn + j];
    q[d * Hn + k] = s;
  }
}

// 5) score[d,t] = V . tanh(q[d] + b2 + y[d,t] @ W2^T) + Vb
__global__ __launch_bounds__(256)
void score_kernel(const float* __restrict__ y, const float* __restrict__ q,
                  const float* __restrict__ W2, const float* __restrict__ b2,
                  const float* __restrict__ Vw, const float* __restrict__ Vb,
                  float* __restrict__ scores)
{
  __shared__ float sW2[Hn * Hn];
  __shared__ float sqb[Hn];
  __shared__ float sv[Hn];
  int d = blockIdx.y, tid = threadIdx.x;
  for (int i = tid; i < Hn * Hn; i += 256) sW2[i] = W2[i];
  if (tid < Hn) { sqb[tid] = q[d * Hn + tid] + b2[tid]; sv[tid] = Vw[tid]; }
  __syncthreads();
  int t = blockIdx.x * 256 + tid;
  const float* yr = y + ((size_t)d * Tn + t) * Hn;
  float yrow[Hn];
#pragma unroll
  for (int j = 0; j < Hn; j += 4) {
    fvec4 p = *(const fvec4*)(yr + j);
    yrow[j] = p[0]; yrow[j + 1] = p[1]; yrow[j + 2] = p[2]; yrow[j + 3] = p[3];
  }
  float acc = 0.f;
  for (int k = 0; k < Hn; ++k) {
    float s = sqb[k];
    const float* wr = sW2 + k * Hn;
#pragma unroll
    for (int j = 0; j < Hn; ++j) s += yrow[j] * wr[j];
    acc += sv[k] * fast_tanh(s);
  }
  scores[d * Tn + t] = acc + Vb[0];
}

// 6) softmax stats per day
__global__ __launch_bounds__(256)
void stats_kernel(const float* __restrict__ scores, float* __restrict__ smax,
                  float* __restrict__ sden)
{
  __shared__ float red[256];
  int d = blockIdx.x, tid = threadIdx.x;
  float m = -1e30f;
  for (int i = tid; i < Tn; i += 256) m = fmaxf(m, scores[d * Tn + i]);
  red[tid] = m; __syncthreads();
  for (int s = 128; s > 0; s >>= 1) { if (tid < s) red[tid] = fmaxf(red[tid], red[tid + s]); __syncthreads(); }
  float mx = red[0]; __syncthreads();
  float sum = 0.f;
  for (int i = tid; i < Tn; i += 256) sum += __expf(scores[d * Tn + i] - mx);
  red[tid] = sum; __syncthreads();
  for (int s = 128; s > 0; s >>= 1) { if (tid < s) red[tid] += red[tid + s]; __syncthreads(); }
  if (tid == 0) { smax[d] = mx; sden[d] = red[0]; }
}

__global__ void zero_kernel(float* p, int n) {
  int i = blockIdx.x * 256 + threadIdx.x;
  if (i < n) p[i] = 0.f;
}

// 7) ctx[d] = sum_t softmax(score)[d,t] * y[d,t,:]   (partial sums + atomics)
__global__ __launch_bounds__(64)
void ctx_kernel(const float* __restrict__ y, const float* __restrict__ scores,
                const float* __restrict__ smax, const float* __restrict__ sden,
                float* __restrict__ ctx)
{
  int d = blockIdx.x, k = threadIdx.x;
  int t0 = blockIdx.y * 512;
  float mx = smax[d], inv = 1.f / sden[d];
  float acc = 0.f;
  for (int t = t0; t < t0 + 512; ++t) {
    float a = __expf(scores[d * Tn + t] - mx);
    acc += a * y[((size_t)d * Tn + t) * Hn + k];
  }
  atomicAdd(&ctx[d * Hn + k], acc * inv);
}

// 8) day LSTM (single step, h0=c0=0) + day attention + stock head
__global__ __launch_bounds__(64)
void final_kernel(const float* __restrict__ ctx,
                  const float* __restrict__ Wih, const float* __restrict__ bih,
                  const float* __restrict__ bhh,
                  const float* __restrict__ dW1, const float* __restrict__ db1,
                  const float* __restrict__ dW2, const float* __restrict__ db2,
                  const float* __restrict__ dV,  const float* __restrict__ dVb,
                  const float* __restrict__ sw,  const float* __restrict__ sb,
                  float* __restrict__ out)
{
  __shared__ float sctx[Dn * Hn], sfull[Dn * Hn], slast[Dn * Hn];
  __shared__ float red[Hn], s2[Dn], aw[Dn];
  int k = threadIdx.x;
  for (int d = 0; d < Dn; ++d) sctx[d * Hn + k] = ctx[d * Hn + k];
  __syncthreads();
  for (int d = 0; d < Dn; ++d) {
    float gi = bih[k] + bhh[k];
    float gg = bih[128 + k] + bhh[128 + k];
    float go = bih[192 + k] + bhh[192 + k];
    for (int j = 0; j < Hn; ++j) {
      float c = sctx[d * Hn + j];
      gi += c * Wih[k * Hn + j];
      gg += c * Wih[(128 + k) * Hn + j];
      go += c * Wih[(192 + k) * Hn + j];
    }
    float cd = fast_sig(gi) * fast_tanh(gg);
    float hd = fast_sig(go) * fast_tanh(cd);
    slast[d * Hn + k] = hd;
    sfull[d * Hn + k] = fast_tanh(hd);
  }
  __syncthreads();
  for (int d = 0; d < Dn; ++d) {
    float s = db1[k] + db2[k];
    for (int j = 0; j < Hn; ++j)
      s += slast[d * Hn + j] * dW1[k * Hn + j] + sfull[d * Hn + j] * dW2[k * Hn + j];
    red[k] = dV[k] * fast_tanh(s);
    __syncthreads();
    for (int st = 32; st > 0; st >>= 1) { if (k < st) red[k] += red[k + st]; __syncthreads(); }
    if (k == 0) s2[d] = red[0] + dVb[0];
    __syncthreads();
  }
  if (k == 0) {
    float mx = -1e30f;
    for (int d = 0; d < Dn; ++d) mx = fmaxf(mx, s2[d]);
    float sum = 0.f;
    for (int d = 0; d < Dn; ++d) { aw[d] = __expf(s2[d] - mx); sum += aw[d]; }
    for (int d = 0; d < Dn; ++d) aw[d] /= sum;
  }
  __syncthreads();
  float ft = 0.f;
  for (int d = 0; d < Dn; ++d) ft += aw[d] * sfull[d * Hn + k];
  red[k] = ft * sw[k];
  __syncthreads();
  for (int st = 32; st > 0; st >>= 1) { if (k < st) red[k] += red[k + st]; __syncthreads(); }
  if (k == 0) out[0] = 3.f * fast_tanh(red[0] + sb[0]);
}

// ---------------------------------------------------------------------------
extern "C" void kernel_launch(void* const* d_in, const int* in_sizes, int n_in,
                              void* d_out, int out_size, void* d_ws, size_t ws_size,
                              hipStream_t stream) {
  (void)in_sizes; (void)n_in; (void)out_size; (void)ws_size;
  const float* x       = (const float*)d_in[0];
  const float* tim     = (const float*)d_in[1];
  // d_in[2] = num_layer (unused)
  const float* W_all_w = (const float*)d_in[3];
  const float* W_all_b = (const float*)d_in[4];
  const float* U_all_w = (const float*)d_in[5];
  const float* U_all_b = (const float*)d_in[6];
  const float* W_d_w   = (const float*)d_in[7];
  const float* W_d_b   = (const float*)d_in[8];
  const float* ta_W1_w = (const float*)d_in[9];
  const float* ta_W1_b = (const float*)d_in[10];
  const float* ta_W2_w = (const float*)d_in[11];
  const float* ta_W2_b = (const float*)d_in[12];
  const float* ta_V_w  = (const float*)d_in[13];
  const float* ta_V_b  = (const float*)d_in[14];
  const float* lstm_Wih= (const float*)d_in[15];
  const float* lstm_bih= (const float*)d_in[17];
  const float* lstm_bhh= (const float*)d_in[18];
  const float* da_W1_w = (const float*)d_in[19];
  const float* da_W1_b = (const float*)d_in[20];
  const float* da_W2_w = (const float*)d_in[21];
  const float* da_W2_b = (const float*)d_in[22];
  const float* da_V_w  = (const float*)d_in[23];
  const float* da_V_b  = (const float*)d_in[24];
  const float* stock_w = (const float*)d_in[25];
  const float* stock_b = (const float*)d_in[26];

  // workspace carve-up (256B aligned)
  char* ws = (char*)d_ws;
  size_t off = 0;
  auto carve = [&](size_t bytes) {
    void* p = ws + off;
    off = (off + bytes + 255) & ~(size_t)255;
    return p;
  };
  unsigned int* pbU  = (unsigned int*)carve(98304u * 4);   // packed U bf16 frags
  unsigned int* pbW  = (unsigned int*)carve(8192u * 4);    // packed W_all
  unsigned int* pbWd = (unsigned int*)carve(2048u * 4);    // packed W_d
  float* ux    = (float*)carve((size_t)Tn * Dn * G4n * 4); // [T][D][256]
  float* ybuf  = (float*)carve((size_t)Dn * Tn * Hn * 4);  // [D][T][64]
  float* hfin  = (float*)carve(Dn * Hn * 4);
  float* qbuf  = (float*)carve(Dn * Hn * 4);
  float* scores= (float*)carve((size_t)Dn * Tn * 4);
  float* smax  = (float*)carve(Dn * 4);
  float* sden  = (float*)carve(Dn * 4);
  float* ctx   = (float*)carve(Dn * Hn * 4);

  pack_weights<<<424, 256, 0, stream>>>(U_all_w, W_all_w, W_d_w, pbU, pbW, pbWd);
  ux_gemm<<<dim3(4, 320), 256, 0, stream>>>(x, pbU, U_all_b, W_all_b, ux);
  time_lstm_scan<<<1, 256, 0, stream>>>(ux, tim, pbW, pbWd, W_d_b, ybuf, hfin);
  q_kernel<<<1, 64, 0, stream>>>(hfin, ta_W1_w, ta_W1_b, qbuf);
  score_kernel<<<dim3(32, Dn), 256, 0, stream>>>(ybuf, qbuf, ta_W2_w, ta_W2_b,
                                                 ta_V_w, ta_V_b, scores);
  stats_kernel<<<Dn, 256, 0, stream>>>(scores, smax, sden);
  zero_kernel<<<2, 256, 0, stream>>>(ctx, Dn * Hn);
  ctx_kernel<<<dim3(Dn, 16), 64, 0, stream>>>(ybuf, scores, smax, sden, ctx);
  final_kernel<<<1, 64, 0, stream>>>(ctx, lstm_Wih, lstm_bih, lstm_bhh,
                                     da_W1_w, da_W1_b, da_W2_w, da_W2_b,
                                     da_V_w, da_V_b, stock_w, stock_b,
                                     (float*)d_out);
}
